// RSSMWorldModel_68169720922585
// MI455X (gfx1250) — matrix operand order repaired and checked by reading, not compiled, not run on hardware
//
#include <hip/hip_runtime.h>
#include <stdint.h>
#include <stddef.h>

// ---------------------------------------------------------------------------
// RSSM observe scan for MI455X (gfx1250, wave32, WMMA bf16 16x16x32).
// 16 persistent blocks x 512 threads (16 waves); block b owns batch rows
// [16b, 16b+16) for all T steps. Weights pre-packed (f32 -> bf16) into WMMA
// B-fragment order in d_ws. Streams (obs in, out) use non-temporal hints so
// packed weights stay L2-resident; next-step obs/act prefetched each step.
// All GEMM loops are software-pipelined one K-tile ahead so WMMA operands
// come from loads issued a full iteration earlier (partial loadcnt waits).
// ---------------------------------------------------------------------------

typedef __attribute__((ext_vector_type(16))) __bf16 v16bf;
typedef __attribute__((ext_vector_type(8)))  float  v8f;
typedef __attribute__((ext_vector_type(4)))  float  v4f;

#define T_     128
#define OBS_   768
#define ACT_   7
#define DETER_ 512
#define HID_   512
#define TOT_   1024
#define K_X    1056           // 1024 stoch + 7 action, padded to 32

#define NKT_IH 33
#define NT_IH  96
#define NKT_HH 16
#define NT_HH  96
#define NKT_P1 16
#define NT_P1  32
#define NKT_P2 16
#define NT_P2  64
#define NKT_Q1 40
#define NT_Q1  32
#define NKT_Q2 16
#define NT_Q2  64

#define FRAG_DW 256           // dwords per packed 32x16 bf16 fragment

#define OFF_IH 0
#define OFF_HH (OFF_IH + NT_IH*NKT_IH*FRAG_DW)
#define OFF_P1 (OFF_HH + NT_HH*NKT_HH*FRAG_DW)
#define OFF_P2 (OFF_P1 + NT_P1*NKT_P1*FRAG_DW)
#define OFF_Q1 (OFF_P2 + NT_P2*NKT_P2*FRAG_DW)
#define OFF_Q2 (OFF_Q1 + NT_Q1*NKT_Q1*FRAG_DW)

#define OUT_F  3584           // deter 512 | stoch 1024 | prior 1024 | post 1024
#define NTHR   512            // 16 waves

__device__ __forceinline__ uint16_t f2bf_bits(float f) {
    __bf16 h = (__bf16)f;
    union { __bf16 b; uint16_t u; } cv; cv.b = h; return cv.u;
}
__device__ __forceinline__ uint32_t pack2bf(float a, float b) {
    return (uint32_t)f2bf_bits(a) | ((uint32_t)f2bf_bits(b) << 16);
}

// Pack W[K x N] (row-major f32) into per-wave-contiguous WMMA B fragments.
// B order used here (bf16 32x16, k contiguous per lane half): VGPR j, lanes 0-15 hold K=2j,2j+1 at col N=lane;
// lanes 16-31 hold K=16+2j,16+2j+1 at col N=lane-16.  Fragment f=(nt*NKT+kt),
// dword index = f*256 + lane*8 + j.  Rows >= K are zero-padded.
__global__ void pack_w(const float* __restrict__ W, uint32_t* __restrict__ dst,
                       int K, int N, int NKT) {
    int idx = blockIdx.x * blockDim.x + threadIdx.x;
    int total = (N >> 4) * NKT * FRAG_DW;
    if (idx >= total) return;
    int j    = idx & 7;
    int lane = (idx >> 3) & 31;
    int frag = idx >> 8;
    int kt = frag % NKT;
    int nt = frag / NKT;
    int n  = (nt << 4) + (lane & 15);
    int k0 = (kt << 5) + (j << 1) + ((lane & 16) ? 16 : 0);
    float a = (k0     < K) ? W[(size_t)k0       * N + n] : 0.f;
    float b = (k0 + 1 < K) ? W[(size_t)(k0 + 1) * N + n] : 0.f;
    dst[idx] = pack2bf(a, b);
}

// A fragment (bf16 16x32) from row-major LDS [16 x stride]:
// lane L (L<16): row M=L, K = kt*32 + {0..15}; lane L+16: {16..31}.
__device__ __forceinline__ v16bf load_a(const __bf16* base, int stride, int kt, int lane) {
    const __bf16* p = base + (lane & 15) * stride + (kt << 5) + ((lane & 16) ? 16 : 0);
    union { uint4 u[2]; v16bf v; } r;
    r.u[0] = *(const uint4*)p;
    r.u[1] = *(const uint4*)(p + 8);
    return r.v;
}

__device__ __forceinline__ v16bf load_b(const uint32_t* __restrict__ ws, int frag, int lane) {
    const uint32_t* p = ws + (size_t)frag * FRAG_DW + (lane << 3);
    union { uint4 u[2]; v16bf v; } r;
    r.u[0] = *(const uint4*)p;
    r.u[1] = *(const uint4*)(p + 4);
    return r.v;
}

__device__ __forceinline__ v8f wmma_bf(v16bf a, v16bf b, v8f c) {
    return __builtin_amdgcn_wmma_f32_16x16x32_bf16(false, a, false, b, (short)0, c, false, false);
}

__device__ __forceinline__ float sigm(float x) { return 1.f / (1.f + __expf(-x)); }
__device__ __forceinline__ float eluf(float x) { return x > 0.f ? x : (__expf(x) - 1.f); }
__device__ __forceinline__ uint32_t hashu(uint32_t x) {
    x ^= x >> 16; x *= 0x7feb352du; x ^= x >> 15; x *= 0x846ca68bu; x ^= x >> 16; return x;
}

// Single-accumulator GEMM tile with B (and A) pipelined one K-tile ahead.
// A is read from one or two LDS regions (deter | obs split for posterior).
__device__ __forceinline__ v8f gemm_tile(const __bf16* a0, int s0, int n0,   // region 0
                                         const __bf16* a1, int s1,           // region 1 (kt>=n0)
                                         const uint32_t* wsb, int fbase, int nkt,
                                         int lane) {
    v8f acc = {};
    v16bf a = load_a(a0, s0, 0, lane);
    v16bf b = load_b(wsb, fbase, lane);
    for (int kt = 0; kt < nkt - 1; ++kt) {
        int kn = kt + 1;
        v16bf a2 = (kn < n0) ? load_a(a0, s0, kn, lane) : load_a(a1, s1, kn - n0, lane);
        v16bf b2 = load_b(wsb, fbase + kn, lane);
        acc = wmma_bf(a, b, acc);
        a = a2; b = b2;
    }
    return wmma_bf(a, b, acc);
}

__global__ __launch_bounds__(NTHR) void rssm_observe(
    const float* __restrict__ obs, const float* __restrict__ act,
    const float* __restrict__ b_ih, const float* __restrict__ b_hh,
    const float* __restrict__ pb1, const float* __restrict__ pb2,
    const float* __restrict__ qb1, const float* __restrict__ qb2,
    const uint32_t* __restrict__ ws, float* __restrict__ out)
{
    extern __shared__ char smem_raw[];
    __bf16* x_bf   = (__bf16*)smem_raw;                 // 16 x 1056 bf16
    __bf16* det_b0 = x_bf   + 16 * K_X;                 // 16 x 512  bf16 (ping)
    __bf16* det_b1 = det_b0 + 16 * DETER_;              // 16 x 512  bf16 (pong)
    __bf16* obs_bf = det_b1 + 16 * DETER_;              // 16 x 768  bf16
    __bf16* h1_bf  = obs_bf + 16 * OBS_;                // 16 x 512  bf16
    float*  det_f  = (float*)(h1_bf + 16 * HID_);       // 16 x 512  f32
    float*  plog   = det_f + 16 * DETER_;               // 16 x 1024 f32

    const int tid  = threadIdx.x;
    const int lane = tid & 31;
    const int wave = tid >> 5;                          // 0..15
    const int m0   = blockIdx.x << 4;                   // global batch-row base

    const uint32_t* wih = ws + OFF_IH;
    const uint32_t* whh = ws + OFF_HH;
    const uint32_t* wp1 = ws + OFF_P1;
    const uint32_t* wp2 = ws + OFF_P2;
    const uint32_t* wq1 = ws + OFF_Q1;
    const uint32_t* wq2 = ws + OFF_Q2;

    // initial state: stoch=0 (x incl. action pad), deter=0
    for (int i = tid; i < 16 * K_X;    i += NTHR) x_bf[i] = (__bf16)0.f;
    for (int i = tid; i < 16 * DETER_; i += NTHR) { det_b0[i] = (__bf16)0.f; det_f[i] = 0.f; }
    __syncthreads();

    for (int t = 0; t < T_; ++t) {
        __bf16* det_old = (t & 1) ? det_b1 : det_b0;
        __bf16* det_new = (t & 1) ? det_b0 : det_b1;

        // ---- prefetch next step's obs/act rows (global_prefetch_b8) ----
        if (t + 1 < T_) {
            if (tid < 384) {                            // 16 rows x 24 x 128B lines
                int r = tid / 24, l = tid % 24;
                __builtin_prefetch(obs + ((size_t)(m0 + r) * T_ + (t + 1)) * OBS_ + l * 32, 0, 1);
            } else if (tid < 400) {
                int r = tid - 384;
                __builtin_prefetch(act + ((size_t)(m0 + r) * T_ + (t + 1)) * ACT_, 0, 1);
            }
        }

        // ---- stage action (bf16) and obs (NT float4 -> packed bf16) ----
        if (tid < 16 * ACT_) {
            int r = tid / ACT_, c = tid % ACT_;
            x_bf[r * K_X + TOT_ + c] = (__bf16)act[((size_t)(m0 + r) * T_ + t) * ACT_ + c];
        }
        for (int i = tid; i < 16 * (OBS_ / 4); i += NTHR) {     // 3072 float4's
            int r = i / (OBS_ / 4), c4 = i % (OBS_ / 4);
            const v4f v = __builtin_nontemporal_load(
                (const v4f*)(obs + ((size_t)(m0 + r) * T_ + t) * OBS_ + c4 * 4));
            uint32_t* dst = (uint32_t*)(obs_bf + r * OBS_ + c4 * 4);
            dst[0] = pack2bf(v.x, v.y);
            dst[1] = pack2bf(v.z, v.w);
        }
        __syncthreads();

        // ---- GRU: each wave owns 2 deter N-tiles; gates fused in registers;
        //      A + 3 B fragments pipelined one K-tile ahead ----
        for (int jj = 0; jj < 2; ++jj) {
            int dt = (wave << 1) + jj;                  // deter tile 0..31
            v8f ar = {}, az = {}, ani = {}, anh = {};
            {   // x @ w_ih (K tiles 0..32) -> ar, az, ani
                const int fR = (dt     ) * NKT_IH;
                const int fZ = (dt + 32) * NKT_IH;
                const int fN = (dt + 64) * NKT_IH;
                v16bf a  = load_a(x_bf, K_X, 0, lane);
                v16bf br = load_b(wih, fR, lane);
                v16bf bz = load_b(wih, fZ, lane);
                v16bf bn = load_b(wih, fN, lane);
                for (int kt = 0; kt < NKT_IH - 1; ++kt) {
                    v16bf a2  = load_a(x_bf, K_X, kt + 1, lane);
                    v16bf br2 = load_b(wih, fR + kt + 1, lane);
                    v16bf bz2 = load_b(wih, fZ + kt + 1, lane);
                    v16bf bn2 = load_b(wih, fN + kt + 1, lane);
                    ar  = wmma_bf(a, br, ar);
                    az  = wmma_bf(a, bz, az);
                    ani = wmma_bf(a, bn, ani);
                    a = a2; br = br2; bz = bz2; bn = bn2;
                }
                ar  = wmma_bf(a, br, ar);
                az  = wmma_bf(a, bz, az);
                ani = wmma_bf(a, bn, ani);
            }
            {   // deter_old @ w_hh (K tiles 0..15) -> ar, az, anh
                const int fR = (dt     ) * NKT_HH;
                const int fZ = (dt + 32) * NKT_HH;
                const int fN = (dt + 64) * NKT_HH;
                v16bf a  = load_a(det_old, DETER_, 0, lane);
                v16bf br = load_b(whh, fR, lane);
                v16bf bz = load_b(whh, fZ, lane);
                v16bf bn = load_b(whh, fN, lane);
                for (int kt = 0; kt < NKT_HH - 1; ++kt) {
                    v16bf a2  = load_a(det_old, DETER_, kt + 1, lane);
                    v16bf br2 = load_b(whh, fR + kt + 1, lane);
                    v16bf bz2 = load_b(whh, fZ + kt + 1, lane);
                    v16bf bn2 = load_b(whh, fN + kt + 1, lane);
                    ar  = wmma_bf(a, br, ar);
                    az  = wmma_bf(a, bz, az);
                    anh = wmma_bf(a, bn, anh);
                    a = a2; br = br2; bz = bz2; bn = bn2;
                }
                ar  = wmma_bf(a, br, ar);
                az  = wmma_bf(a, bz, az);
                anh = wmma_bf(a, bn, anh);
            }
            int ncol = (dt << 4) + (lane & 15);
            float br_ = b_ih[ncol]              + b_hh[ncol];
            float bz_ = b_ih[DETER_ + ncol]     + b_hh[DETER_ + ncol];
            float bni = b_ih[2 * DETER_ + ncol];
            float bnh = b_hh[2 * DETER_ + ncol];
            #pragma unroll
            for (int g = 0; g < 8; ++g) {
                int m = g + ((lane >> 4) << 3);         // C layout: M=g (+8 hi half)
                float r = sigm(ar[g] + br_);
                float z = sigm(az[g] + bz_);
                float n = tanhf(ani[g] + bni + r * (anh[g] + bnh));
                float dold = det_f[m * DETER_ + ncol];
                float dnew = (1.f - z) * n + z * dold;
                det_f[m * DETER_ + ncol]   = dnew;      // same-thread RAW -> safe
                det_new[m * DETER_ + ncol] = (__bf16)dnew;
                __builtin_nontemporal_store(dnew,
                    &out[((size_t)(m0 + m) * T_ + t) * OUT_F + ncol]);
            }
        }
        __syncthreads();

        // ---- prior hidden: h1 = elu(deter @ W1 + b1) ----
        for (int jj = 0; jj < 2; ++jj) {
            int nt = (wave << 1) + jj;
            v8f acc = gemm_tile(det_new, DETER_, NKT_P1, det_new, DETER_,
                                wp1, nt * NKT_P1, NKT_P1, lane);
            int ncol = (nt << 4) + (lane & 15);
            float b1 = pb1[ncol];
            #pragma unroll
            for (int g = 0; g < 8; ++g) {
                int m = g + ((lane >> 4) << 3);
                h1_bf[m * HID_ + ncol] = (__bf16)eluf(acc[g] + b1);
            }
        }
        __syncthreads();

        // ---- prior logits: h1 @ W2 + b2 -> out[1536..2560) ----
        for (int jj = 0; jj < 4; ++jj) {
            int nt = (wave << 2) + jj;
            v8f acc = gemm_tile(h1_bf, HID_, NKT_P2, h1_bf, HID_,
                                wp2, nt * NKT_P2, NKT_P2, lane);
            int ncol = (nt << 4) + (lane & 15);
            float b2 = pb2[ncol];
            #pragma unroll
            for (int g = 0; g < 8; ++g) {
                int m = g + ((lane >> 4) << 3);
                __builtin_nontemporal_store(acc[g] + b2,
                    &out[((size_t)(m0 + m) * T_ + t) * OUT_F + 1536 + ncol]);
            }
        }
        __syncthreads();                                // h1_bf reuse hazard

        // ---- posterior hidden: elu([deter|obs] @ W1 + b1), split-K A regions ----
        for (int jj = 0; jj < 2; ++jj) {
            int nt = (wave << 1) + jj;
            v8f acc = gemm_tile(det_new, DETER_, 16, obs_bf, OBS_,
                                wq1, nt * NKT_Q1, NKT_Q1, lane);
            int ncol = (nt << 4) + (lane & 15);
            float b1 = qb1[ncol];
            #pragma unroll
            for (int g = 0; g < 8; ++g) {
                int m = g + ((lane >> 4) << 3);
                h1_bf[m * HID_ + ncol] = (__bf16)eluf(acc[g] + b1);
            }
        }
        __syncthreads();

        // ---- posterior logits -> LDS (for sampling) + out[2560..3584) ----
        for (int jj = 0; jj < 4; ++jj) {
            int nt = (wave << 2) + jj;
            v8f acc = gemm_tile(h1_bf, HID_, NKT_Q2, h1_bf, HID_,
                                wq2, nt * NKT_Q2, NKT_Q2, lane);
            int ncol = (nt << 4) + (lane & 15);
            float b2 = qb2[ncol];
            #pragma unroll
            for (int g = 0; g < 8; ++g) {
                int m = g + ((lane >> 4) << 3);
                float v = acc[g] + b2;
                plog[m * TOT_ + ncol] = v;
                __builtin_nontemporal_store(v,
                    &out[((size_t)(m0 + m) * T_ + t) * OUT_F + 2560 + ncol]);
            }
        }
        __syncthreads();

        // ---- softmax + categorical sample; stoch = one-hot (ST fwd value) ----
        {
            int row = tid >> 5, gi = tid & 31;          // 512 dists, 1 per thread
            const float* lg = plog + row * TOT_ + (gi << 5);
            float mx = lg[0];
            for (int c = 1; c < 32; ++c) mx = fmaxf(mx, lg[c]);
            float s = 0.f;
            for (int c = 0; c < 32; ++c) s += __expf(lg[c] - mx);
            uint32_t h = hashu(((uint32_t)(m0 + row) * 131071u) ^
                               ((uint32_t)gi * 8191u) ^ ((uint32_t)t * 524287u) ^ 0x9E3779B9u);
            float u = (float)(h & 0xFFFFFFu) * (1.f / 16777216.f);
            float target = u * s, csum = 0.f;
            int pick = 31;
            for (int c = 0; c < 32; ++c) {
                csum += __expf(lg[c] - mx);
                if (csum >= target) { pick = c; break; }
            }
            __bf16* xrow = x_bf + row * K_X + (gi << 5);
            float*  srow = out + ((size_t)(m0 + row) * T_ + t) * OUT_F + DETER_ + (gi << 5);
            for (int c = 0; c < 32; ++c) {
                float v = (c == pick) ? 1.f : 0.f;
                xrow[c] = (__bf16)v;
                __builtin_nontemporal_store(v, &srow[c]);
            }
        }
        __syncthreads();
    }
}

extern "C" void kernel_launch(void* const* d_in, const int* in_sizes, int n_in,
                              void* d_out, int out_size, void* d_ws, size_t ws_size,
                              hipStream_t stream) {
    (void)in_sizes; (void)n_in; (void)out_size; (void)ws_size;
    const float* obs  = (const float*)d_in[0];
    const float* actp = (const float*)d_in[1];
    const float* w_ih = (const float*)d_in[2];
    const float* w_hh = (const float*)d_in[3];
    const float* b_ih = (const float*)d_in[4];
    const float* b_hh = (const float*)d_in[5];
    const float* pw1  = (const float*)d_in[6];
    const float* pb1  = (const float*)d_in[7];
    const float* pw2  = (const float*)d_in[8];
    const float* pb2  = (const float*)d_in[9];
    const float* qw1  = (const float*)d_in[10];
    const float* qb1  = (const float*)d_in[11];
    const float* qw2  = (const float*)d_in[12];
    const float* qb2  = (const float*)d_in[13];
    uint32_t* ws = (uint32_t*)d_ws;
    float* out = (float*)d_out;

    auto pack = [&](const float* W, int off, int K, int N, int NKT) {
        int total = (N >> 4) * NKT * FRAG_DW;
        pack_w<<<(total + 255) / 256, 256, 0, stream>>>(W, ws + off, K, N, NKT);
    };
    pack(w_ih, OFF_IH, 1031, 1536, NKT_IH);
    pack(w_hh, OFF_HH,  512, 1536, NKT_HH);
    pack(pw1,  OFF_P1,  512,  512, NKT_P1);
    pack(pw2,  OFF_P2,  512, 1024, NKT_P2);
    pack(qw1,  OFF_Q1, 1280,  512, NKT_Q1);
    pack(qw2,  OFF_Q2,  512, 1024, NKT_Q2);

    size_t smem = (size_t)16 * K_X * 2         // x bf16
                + (size_t)2 * 16 * DETER_ * 2  // deter ping/pong bf16
                + (size_t)16 * OBS_ * 2        // obs bf16
                + (size_t)16 * HID_ * 2        // h1 bf16
                + (size_t)16 * DETER_ * 4      // deter f32
                + (size_t)16 * TOT_ * 4;       // post logits f32  (= 205824 B)
    hipFuncSetAttribute((const void*)rssm_observe,
                        hipFuncAttributeMaxDynamicSharedMemorySize, (int)smem);
    rssm_observe<<<16, NTHR, smem, stream>>>(obs, actp, b_ih, b_hh,
                                             pb1, pb2, qb1, qb2, ws, out);
}
